// LQSharedFFModel_49306224558577
// MI455X (gfx1250) — compile-verified, matrix operand order checked by reading, not run on hardware
//
#include <hip/hip_runtime.h>
#include <hip/hip_bf16.h>

typedef __attribute__((ext_vector_type(16))) __bf16 v16bf;
typedef __attribute__((ext_vector_type(8)))  float  v8f;

#define DT_F     0.01f
#define EM2_F    0.98f
#define NT_STEPS 100
#define HIST     50
#define MT       16          // batch rows per workgroup
#define DDIM     64
#define DPDIM    32
#define KIN      704         // 11*64 state K (time col folded into bias)
#define KT1      22          // 704/32
#define KT2      16          // 512/32
#define KT3      16
#define H1S      528         // padded bf16 row stride for activations

// ---- LDS layout (bytes), all 16B aligned ----
#define OFF_XR   0
#define SZ_XR    (MT*HIST*DDIM*2)        // 102400
#define OFF_H1   (OFF_XR + SZ_XR)
#define SZ_H     (MT*H1S*2)              // 16896
#define OFF_H2   (OFF_H1 + SZ_H)
#define OFF_A1   (OFF_H2 + SZ_H)
#define SZ_M64   (64*65*4)               // padded 64x64 f32
#define OFF_A2   (OFF_A1 + SZ_M64)
#define OFF_A3   (OFF_A2 + SZ_M64)
#define OFF_Q    (OFF_A3 + SZ_M64)
#define OFF_G    (OFF_Q  + SZ_M64)
#define OFF_SIG  (OFF_G  + SZ_M64)
#define OFF_BM   (OFF_SIG+ SZ_M64)
#define SZ_BM    (64*33*4)
#define OFF_R    (OFF_BM + SZ_BM)
#define SZ_R     (32*33*4)
#define OFF_X    (OFF_R  + SZ_R)
#define SZ_V     (MT*DDIM*4)
#define OFF_Y    (OFF_X  + SZ_V)
#define OFF_XC   (OFF_Y  + SZ_V)
#define OFF_Z    (OFF_XC + SZ_V)
#define OFF_DW   (OFF_Z  + SZ_V)
#define OFF_PI   (OFF_DW + SZ_V)
#define SZ_PI    (MT*DPDIM*4)
#define SMEM_BYTES (OFF_PI + SZ_PI)      // 271232 B

// packed weight sizes (dwords)
#define P1_DW (32*KT1*32*8)   // 180224
#define P2_DW (32*KT2*32*8)   // 131072
#define P3_DW ( 2*KT3*32*8)   //   8192

// ---------------- weight pre-pack: f32 -> bf16 WMMA B-fragments ----------------
// Fragment layout per (ntile, ktile): dword index = lane*8 + v, where
// n = ntile*16 + lane%16 ; koff = lane<16 ? 0 : 8 ; k = ktile*32 + koff + (v<4 ? 2v : 2v+8)
// dword packs bf16(W[k][n]) | bf16(W[k+1][n]) << 16.
// W1 rows are permuted: logical k' = s*64+d  maps to original row d*11+s.
__global__ __launch_bounds__(256) void pack_weights(const float* __restrict__ W1,
                                                    const float* __restrict__ W2,
                                                    const float* __restrict__ W3,
                                                    unsigned* __restrict__ P1,
                                                    unsigned* __restrict__ P2,
                                                    unsigned* __restrict__ P3) {
  int idx = blockIdx.x * 256 + threadIdx.x;
  const float* W; unsigned* P; int ntK, ncols, rem; int permute = 0;
  if (idx < P1_DW)                { W = W1; P = P1; ntK = KT1; ncols = 512; rem = idx; permute = 1; }
  else if (idx < P1_DW + P2_DW)   { W = W2; P = P2; ntK = KT2; ncols = 512; rem = idx - P1_DW; }
  else if (idx < P1_DW + P2_DW + P3_DW) { W = W3; P = P3; ntK = KT3; ncols = 32; rem = idx - P1_DW - P2_DW; }
  else return;
  int v    = rem & 7;
  int lane = (rem >> 3) & 31;
  int blk  = rem >> 8;            // ntg*ntK + kt
  int kt   = blk % ntK;
  int ntg  = blk / ntK;
  int n    = ntg * 16 + (lane & 15);
  int koff = (lane < 16) ? 0 : 8;
  int k    = kt * 32 + koff + ((v < 4) ? 2 * v : 2 * v + 8);
  int r0 = k, r1 = k + 1;
  if (permute) {                  // k' = s*64 + d  ->  orig row d*11 + s
    int s0 = r0 >> 6, d0 = r0 & 63; r0 = d0 * 11 + s0;
    int s1 = r1 >> 6, d1 = r1 & 63; r1 = d1 * 11 + s1;
  }
  union { __bf16 h[2]; unsigned u; } pk;
  pk.h[0] = (__bf16)W[r0 * ncols + n];
  pk.h[1] = (__bf16)W[r1 * ncols + n];
  P[rem] = pk.u;
}

__device__ __forceinline__ v8f wmma_bf16(uint4 a0, uint4 a1, uint4 b0, uint4 b1, v8f c) {
  union { uint4 q[2]; v16bf v; } A, B;
  A.q[0] = a0; A.q[1] = a1; B.q[0] = b0; B.q[1] = b1;
  return __builtin_amdgcn_wmma_f32_16x16x32_bf16(false, A.v, false, B.v, (short)0, c, false, false);
}

// ---------------- main fused time-marching kernel ----------------
__global__ __launch_bounds__(256) void lq_main(
    const float* __restrict__ dw,   const float* __restrict__ x_init, const float* __restrict__ wgt,
    const float* __restrict__ A1g,  const float* __restrict__ A2g,   const float* __restrict__ A3g,
    const float* __restrict__ Bg,   const float* __restrict__ SIGg,  const float* __restrict__ Qg,
    const float* __restrict__ Rg,   const float* __restrict__ Gg,
    const float* __restrict__ W1g,  const float* __restrict__ b1g,
    const float* __restrict__ b2g,  const float* __restrict__ b3g,
    const unsigned* __restrict__ P1, const unsigned* __restrict__ P2, const unsigned* __restrict__ P3,
    float* __restrict__ out) {
  extern __shared__ char smem[];
  __bf16* XR  = (__bf16*)(smem + OFF_XR);   // [MT][HIST][64] ring, bf16
  __bf16* H1  = (__bf16*)(smem + OFF_H1);   // [MT][H1S]
  __bf16* H2  = (__bf16*)(smem + OFF_H2);
  float*  A1s = (float*)(smem + OFF_A1);    // padded stride 65
  float*  A2s = (float*)(smem + OFF_A2);
  float*  A3s = (float*)(smem + OFF_A3);
  float*  Qs  = (float*)(smem + OFF_Q);
  float*  Gs  = (float*)(smem + OFF_G);
  float*  SIGs= (float*)(smem + OFF_SIG);
  float*  Bms = (float*)(smem + OFF_BM);    // stride 33
  float*  Rs  = (float*)(smem + OFF_R);     // stride 33
  float*  Xs  = (float*)(smem + OFF_X);     // [MT][64]
  float*  Ys  = (float*)(smem + OFF_Y);
  float*  XCs = (float*)(smem + OFF_XC);
  float*  Zs  = (float*)(smem + OFF_Z);
  float*  DWs = (float*)(smem + OFF_DW);
  float*  PIs = (float*)(smem + OFF_PI);    // [MT][32] f32

  const int tid  = threadIdx.x;
  const int b0   = blockIdx.x * MT;
  const int lane = tid & 31;
  const int wv   = tid >> 5;

  // ---- stage small matrices (padded to avoid bank conflicts) ----
  for (int i = tid; i < 64 * 64; i += 256) {
    int rr = i >> 6, cc = i & 63, p = rr * 65 + cc;
    A1s[p] = A1g[i]; A2s[p] = A2g[i]; A3s[p] = A3g[i];
    Qs[p] = Qg[i];   Gs[p] = Gg[i];   SIGs[p] = SIGg[i];
  }
  for (int i = tid; i < 64 * 32; i += 256) Bms[(i >> 5) * 33 + (i & 31)] = Bg[i];
  for (int i = tid; i < 32 * 32; i += 256) Rs [(i >> 5) * 33 + (i & 31)] = Rg[i];

  // ---- per-thread scalar state: fixed d, 4 rows r0+4i ----
  const int d  = tid & 63;
  const int r0 = tid >> 6;
  const float E49 = __powf(EM2_F, 49.0f);
  float xq[4], Sq[4], gq[4], w49q[4], rewp[4], xcq[4];
  float dec = 1.0f;
#pragma unroll
  for (int i = 0; i < 4; ++i) {
    int r = r0 + 4 * i;
    const float* xi = x_init + ((size_t)(b0 + r) * 64 + d) * HIST;
    const float* wi = wgt    + ((size_t)(b0 + r) * 64 + d) * HIST;
    float s = 0.f;
    for (int j = 0; j < HIST; ++j) {
      XR[(r * HIST + j) * 64 + d] = (__bf16)xi[j];
      s += wi[j];
    }
    Sq[i] = s; xq[i] = xi[HIST - 1];
    gq[i] = wi[0]; w49q[i] = wi[HIST - 1]; rewp[i] = 0.f;
  }
  __syncthreads();

  for (int t = 0; t <= NT_STEPS; ++t) {
    const int rbase = t % HIST;
    // ======= phase A: shift / y / zeta (scalar, own (r,d)) =======
    if (t > 0) {
      const int ob = (t - 1) % HIST;
      dec *= EM2_F;
#pragma unroll
      for (int i = 0; i < 4; ++i) {
        int r = r0 + 4 * i;
        Sq[i] = (Sq[i] - gq[i]) * EM2_F + xq[i];       // pop oldest, decay, append x
        XR[(r * HIST + ob) * 64 + d] = (__bf16)xq[i];  // push x into ring
        float gnew = (t <= HIST - 1)
            ? wgt[((size_t)(b0 + r) * 64 + d) * HIST + t] * dec
            : (float)XR[(r * HIST + rbase) * 64 + d] * E49;
        gq[i] = gnew;                                   // post-shift oldest wh value
        float y = (Sq[i] - 0.5f * (gnew + xq[i])) * DT_F;
        Ys[r * 64 + d] = y;
        Zs[r * 64 + d] = (float)XR[(r * HIST + rbase) * 64 + d];
        Xs[r * 64 + d] = xq[i];
      }
    } else {
#pragma unroll
      for (int i = 0; i < 4; ++i) {
        int r = r0 + 4 * i;
        float y = (Sq[i] - 0.5f * (gq[i] + w49q[i])) * DT_F;
        Ys[r * 64 + d] = y;
        Zs[r * 64 + d] = (float)XR[(r * HIST + 0) * 64 + d];
        Xs[r * 64 + d] = xq[i];
      }
    }
    if (t < NT_STEPS) {
#pragma unroll
      for (int i = 0; i < 4; ++i) {
        int r = r0 + 4 * i;
        DWs[r * 64 + d] = dw[((size_t)(b0 + r) * 64 + d) * NT_STEPS + t];
      }
    }
    __syncthreads();
    // ======= phase B: x_common = x + y @ A3 =======
#pragma unroll
    for (int i = 0; i < 4; ++i) {
      int r = r0 + 4 * i;
      float acc = 0.f;
#pragma unroll 8
      for (int k = 0; k < 64; ++k) acc += Ys[r * 64 + k] * A3s[k * 65 + d];
      float xc = xq[i] + acc;
      XCs[r * 64 + d] = xc; xcq[i] = xc;
    }
    __syncthreads();
    if (t == NT_STEPS) {  // terminal reward: xc @ G xc
#pragma unroll
      for (int i = 0; i < 4; ++i) {
        int r = r0 + 4 * i;
        float qv = 0.f;
#pragma unroll 8
        for (int k = 0; k < 64; ++k) qv += XCs[r * 64 + k] * Gs[k * 65 + d];
        rewp[i] += xcq[i] * qv;
      }
      break;
    }
    // ======= phase C: MLP via bf16 WMMA =======
    {
      const float tscale = (float)t * DT_F;
      const int m    = lane & 15;
      const int koff = (lane < 16) ? 0 : 8;
      const int mb   = (lane < 16) ? 0 : 8;
      // ---- layer 1: state(16x704) @ W1 -> H1 ----
      v8f acc[4];
#pragma unroll
      for (int q = 0; q < 4; ++q) {
#pragma unroll
        for (int e = 0; e < 8; ++e) acc[q][e] = 0.f;
      }
      for (int kt = 0; kt < KT1; ++kt) {
        int K1 = kt * 32 + koff;
        int srel = K1 >> 6, dd = K1 & 63;
        int ps = rbase + 39 + srel; if (ps >= HIST) ps -= HIST;
        const uint4* ap = (const uint4*)(XR + (m * HIST + ps) * 64 + dd);
        uint4 a0 = ap[0], a1 = ap[2];  // dd..dd+7 and dd+16..dd+23
#pragma unroll
        for (int nt = 0; nt < 4; ++nt) {
          int ntg = wv * 4 + nt;
          const uint4* bp = (const uint4*)(P1 + ((ntg * KT1 + kt) * 32 + lane) * 8);
          acc[nt] = wmma_bf16(a0, a1, bp[0], bp[1], acc[nt]);
        }
      }
#pragma unroll
      for (int nt = 0; nt < 4; ++nt) {
        int n = (wv * 4 + nt) * 16 + (lane & 15);
        float bias = b1g[n] + tscale * W1g[704 * 512 + n];
#pragma unroll
        for (int vr = 0; vr < 8; ++vr)
          H1[(vr + mb) * H1S + n] = (__bf16)fmaxf(acc[nt][vr] + bias, 0.f);
      }
      __syncthreads();
      // ---- layer 2: H1(16x512) @ W2 -> H2 ----
#pragma unroll
      for (int q = 0; q < 4; ++q) {
#pragma unroll
        for (int e = 0; e < 8; ++e) acc[q][e] = 0.f;
      }
      for (int kt = 0; kt < KT2; ++kt) {
        int K1 = kt * 32 + koff;
        const uint4* ap = (const uint4*)(H1 + m * H1S + K1);
        uint4 a0 = ap[0], a1 = ap[2];
#pragma unroll
        for (int nt = 0; nt < 4; ++nt) {
          int ntg = wv * 4 + nt;
          const uint4* bp = (const uint4*)(P2 + ((ntg * KT2 + kt) * 32 + lane) * 8);
          acc[nt] = wmma_bf16(a0, a1, bp[0], bp[1], acc[nt]);
        }
      }
#pragma unroll
      for (int nt = 0; nt < 4; ++nt) {
        int n = (wv * 4 + nt) * 16 + (lane & 15);
        float bias = b2g[n];
#pragma unroll
        for (int vr = 0; vr < 8; ++vr)
          H2[(vr + mb) * H1S + n] = (__bf16)fmaxf(acc[nt][vr] + bias, 0.f);
      }
      __syncthreads();
      // ---- layer 3: H2(16x512) @ W3 -> PIs (waves 0,1) ----
      if (wv < 2) {
        v8f a3c;
#pragma unroll
        for (int e = 0; e < 8; ++e) a3c[e] = 0.f;
        for (int kt = 0; kt < KT3; ++kt) {
          int K1 = kt * 32 + koff;
          const uint4* ap = (const uint4*)(H2 + m * H1S + K1);
          const uint4* bp = (const uint4*)(P3 + ((wv * KT3 + kt) * 32 + lane) * 8);
          a3c = wmma_bf16(ap[0], ap[2], bp[0], bp[1], a3c);
        }
        int n = wv * 16 + (lane & 15);
        float bias = b3g[n];
#pragma unroll
        for (int vr = 0; vr < 8; ++vr)
          PIs[(vr + mb) * DPDIM + n] = a3c[vr] + bias;
      }
      __syncthreads();
    }
    // ======= phase D: reward, dx, x update =======
#pragma unroll
    for (int i = 0; i < 4; ++i) {
      int r = r0 + 4 * i;
      float qv = 0.f;
#pragma unroll 8
      for (int k = 0; k < 64; ++k) qv += XCs[r * 64 + k] * Qs[k * 65 + d];
      float rp = xcq[i] * qv;
      if (d < DPDIM) {
        float pv = 0.f;
#pragma unroll 8
        for (int k = 0; k < DPDIM; ++k) pv += PIs[r * DPDIM + k] * Rs[k * 33 + d];
        rp += PIs[r * DPDIM + d] * pv;
      }
      rewp[i] += rp * DT_F;
      float dx = 0.f, nw = 0.f;
#pragma unroll 8
      for (int k = 0; k < 64; ++k) dx += Xs[r * 64 + k] * A1s[d * 65 + k];
#pragma unroll 8
      for (int k = 0; k < 64; ++k) dx += Ys[r * 64 + k] * A2s[d * 65 + k];
#pragma unroll 8
      for (int k = 0; k < 64; ++k) dx += Zs[r * 64 + k] * A3s[d * 65 + k];
#pragma unroll 8
      for (int k = 0; k < DPDIM; ++k) dx += PIs[r * DPDIM + k] * Bms[d * 33 + k];
#pragma unroll 8
      for (int k = 0; k < 64; ++k) nw += DWs[r * 64 + k] * SIGs[d * 65 + k];
      xq[i] += dx * DT_F + nw;
    }
    __syncthreads();
  }
  // ======= per-row reduction over d =======
  __syncthreads();
  if (tid < MT) PIs[tid] = 0.f;
  __syncthreads();
#pragma unroll
  for (int i = 0; i < 4; ++i) atomicAdd(&PIs[r0 + 4 * i], rewp[i]);
  __syncthreads();
  if (tid < MT) out[b0 + tid] = PIs[tid];
}

extern "C" void kernel_launch(void* const* d_in, const int* in_sizes, int n_in,
                              void* d_out, int out_size, void* d_ws, size_t ws_size,
                              hipStream_t stream) {
  const float* dw     = (const float*)d_in[0];
  const float* x_init = (const float*)d_in[1];
  const float* wgt    = (const float*)d_in[2];
  const float* A1     = (const float*)d_in[3];
  const float* A2     = (const float*)d_in[4];
  const float* A3     = (const float*)d_in[5];
  const float* Bm     = (const float*)d_in[6];
  const float* sigma  = (const float*)d_in[7];
  const float* Q      = (const float*)d_in[8];
  const float* R      = (const float*)d_in[9];
  const float* G      = (const float*)d_in[10];
  const float* W1     = (const float*)d_in[11];
  const float* b1     = (const float*)d_in[12];
  const float* W2     = (const float*)d_in[13];
  const float* b2     = (const float*)d_in[14];
  const float* W3     = (const float*)d_in[15];
  const float* b3     = (const float*)d_in[16];

  unsigned* P1 = (unsigned*)d_ws;
  unsigned* P2 = P1 + P1_DW;
  unsigned* P3 = P2 + P2_DW;

  const int total_pack = P1_DW + P2_DW + P3_DW;
  pack_weights<<<(total_pack + 255) / 256, 256, 0, stream>>>(W1, W2, W3, P1, P2, P3);

  (void)hipFuncSetAttribute((const void*)lq_main,
                            hipFuncAttributeMaxDynamicSharedMemorySize, SMEM_BYTES);
  lq_main<<<4096 / MT, 256, SMEM_BYTES, stream>>>(
      dw, x_init, wgt, A1, A2, A3, Bm, sigma, Q, R, G,
      W1, b1, b2, b3, P1, P2, P3, (float*)d_out);
}
